// TransformerBlockClassical_65481071395732
// MI455X (gfx1250) — compile-verified, matrix operand order checked by reading, not run on hardware
//
#include <hip/hip_runtime.h>

typedef __attribute__((ext_vector_type(16))) _Float16 v16h;
typedef __attribute__((ext_vector_type(8)))  float    v8f;
typedef __attribute__((ext_vector_type(4)))  float    f32x4;
typedef int v4i __attribute__((vector_size(16)));

#define EMBED 1024
#define QKV3  3072
#define NHEAD 16
#define DK    64
#define FFN   4096
#define SEQ   2048
#define MTOT  4096   // B*S = 2*2048

// Async global->LDS path (CDNA5 GLOBAL_LOAD_ASYNC_TO_LDS_B128, ASYNCcnt)
#if defined(__HIP_DEVICE_COMPILE__) && __has_builtin(__builtin_amdgcn_global_load_async_to_lds_b128)
#define USE_ASYNC_LDS 1
#else
#define USE_ASYNC_LDS 0
#endif

// Tensor Data Mover path (CDNA5 TENSOR_LOAD_TO_LDS, TENSORcnt)
#if defined(__HIP_DEVICE_COMPILE__) && __has_builtin(__builtin_amdgcn_tensor_load_to_lds)
#define USE_TDM 1
#else
#define USE_TDM 0
#endif

#if USE_ASYNC_LDS
typedef __attribute__((address_space(1))) v4i* gptr_b128;   // global b128 ptr
typedef __attribute__((address_space(3))) v4i* lptr_b128;   // LDS b128 ptr
__device__ __forceinline__ void async_copy_b128(const void* g, void* l) {
    __builtin_amdgcn_global_load_async_to_lds_b128((gptr_b128)(g), (lptr_b128)(l), 0, 0);
}
__device__ __forceinline__ void wait_async0() {
    asm volatile("s_wait_asynccnt 0x0" ::: "memory");
}
#endif

#if USE_TDM
typedef unsigned int u32x4 __attribute__((vector_size(16)));
typedef int          i32x8 __attribute__((vector_size(32)));
typedef int          i32x4 __attribute__((vector_size(16)));

__device__ __forceinline__ void tdm_wait0() {
#if __has_builtin(__builtin_amdgcn_s_wait_tensorcnt)
    __builtin_amdgcn_s_wait_tensorcnt(0);
#else
    asm volatile("s_wait_tensorcnt 0x0" ::: "memory");
#endif
}

// TDM 2D tile load: 128 rows x 32 halfs (64B/row) from row-major global
// (row stride K halfs) into LDS rows padded to 40 halfs (pad 16B per 64B).
__device__ __forceinline__ void tdm_load_tile(const _Float16* gsrc, void* lds, int K) {
    unsigned la = (unsigned)(size_t)(__attribute__((address_space(3))) void*)lds;
    unsigned long long ga = (unsigned long long)(size_t)gsrc;
    u32x4 g0 = { 1u,                                            // count=1 (valid user D#)
                 la,                                            // lds_addr (bytes)
                 (unsigned)ga,                                  // global_addr[31:0]
                 (unsigned)((ga >> 32) & 0x1FFFFFFu) | (2u << 30) }; // addr[56:32] | type=2
    i32x8 g1;
    g1[0] = (int)((1u << 16) |      // data_size = 2 bytes
                  (1u << 20) |      // pad_enable
                  (3u << 22) |      // pad_interval: 16 DWORDs (one 64B row)
                  (3u << 25));      // pad_amount: 4 DWORDs (16B pad)
    g1[1] = (int)0xFFFF0000u;       // tensor_dim0[15:0]=0xFFFF (huge -> no OOB clip)
    g1[2] = (int)0xFFFF7FFFu;       // tensor_dim0 hi=0x7FFF | tensor_dim1 lo=0xFFFF
    g1[3] = (int)((32u << 16) | 0x7FFFu); // tensor_dim1 hi | tile_dim0 = 32 halfs
    g1[4] = 128;                    // tile_dim1 = 128 rows, tile_dim2 = 0
    g1[5] = K;                      // tensor_dim0_stride (halfs), low 32
    g1[6] = 0;                      // stride hi | tensor_dim1_stride lo
    g1[7] = 0;
    i32x4 gz4 = { 0, 0, 0, 0 };
    i32x8 gz8 = { 0, 0, 0, 0, 0, 0, 0, 0 };
    __builtin_amdgcn_tensor_load_to_lds(g0, g1, gz4, gz4, gz8, 0);
}
#endif

// ---------------------------------------------------------------------------
// Fragment loader: v16h with elements 0..7 at p, 8..15 at p+16 (halfs).
// ---------------------------------------------------------------------------
__device__ __forceinline__ v16h load_frag16(const _Float16* p) {
    union { v16h v; f32x4 q[2]; } u;
    u.q[0] = *(const f32x4*)(p);
    u.q[1] = *(const f32x4*)(p + 16);
    return u.v;
}

// ---------------------------------------------------------------------------
// fp32 -> fp16 elementwise conversion (weights)
// ---------------------------------------------------------------------------
__global__ __launch_bounds__(256) void f32_to_f16_kernel(const float* __restrict__ in,
                                                         _Float16* __restrict__ out, int n) {
    int i = blockIdx.x * 256 + threadIdx.x;
    if (i < n) out[i] = (_Float16)in[i];
}

// concat bq|bk|bv -> 3072 fp32
__global__ __launch_bounds__(256) void concat_bias_kernel(const float* __restrict__ a,
                                                          const float* __restrict__ b,
                                                          const float* __restrict__ c,
                                                          float* __restrict__ out) {
    int i = blockIdx.x * 256 + threadIdx.x;
    if (i < QKV3) {
        float v = (i < EMBED) ? a[i] : (i < 2 * EMBED ? b[i - EMBED] : c[i - 2 * EMBED]);
        out[i] = v;
    }
}

// ---------------------------------------------------------------------------
// LayerNorm: one row (1024 fp32) per workgroup (256 thr = 8 waves), f16 out
// ---------------------------------------------------------------------------
__global__ __launch_bounds__(256) void layernorm_f16_kernel(const float* __restrict__ x,
                                                            const float* __restrict__ g,
                                                            const float* __restrict__ b,
                                                            _Float16* __restrict__ out) {
    const int row = blockIdx.x;
    const int t = threadIdx.x;
    const int wave = t >> 5, lane = t & 31;
    const float* xr = x + (size_t)row * EMBED;

    float vals[4];
    float s1 = 0.f, s2 = 0.f;
#pragma unroll
    for (int i = 0; i < 4; ++i) {
        float v = xr[t + i * 256];
        vals[i] = v; s1 += v; s2 += v * v;
    }
#pragma unroll
    for (int off = 1; off < 32; off <<= 1) {
        s1 += __shfl_xor(s1, off, 32);
        s2 += __shfl_xor(s2, off, 32);
    }
    __shared__ float rs1[8], rs2[8];
    if (lane == 0) { rs1[wave] = s1; rs2[wave] = s2; }
    __syncthreads();
    float t1 = 0.f, t2 = 0.f;
#pragma unroll
    for (int w = 0; w < 8; ++w) { t1 += rs1[w]; t2 += rs2[w]; }
    const float mu   = t1 * (1.0f / EMBED);
    const float var  = t2 * (1.0f / EMBED) - mu * mu;
    const float rstd = rsqrtf(var + 1e-5f);
#pragma unroll
    for (int i = 0; i < 4; ++i) {
        int c = t + i * 256;
        out[(size_t)row * EMBED + c] = (_Float16)((vals[i] - mu) * rstd * g[c] + b[c]);
    }
}

// ---------------------------------------------------------------------------
// WMMA GEMM:  out[M,N] = A[M,K] (f16, row-major) x W[N,K]^T (f16) + bias
//   RELU: relu + f16 output;  RESID: add fp32 residual, fp32 output
// 256 threads (8 waves); macro tile 128x128; each wave 32x64.
// Double-buffered LDS; staging via TDM (wave0 issues DMA) when available,
// else per-lane async-to-LDS, else sync. LDS rows padded to 40 halfs =>
// conflict-free ds_load_b128 fragment reads.
// ---------------------------------------------------------------------------
template <bool RELU, bool RESID>
__global__ __launch_bounds__(256) void wmma_gemm_kernel(const _Float16* __restrict__ A,
                                                        const _Float16* __restrict__ W,
                                                        const float* __restrict__ bias,
                                                        const float* __restrict__ resid,
                                                        void* __restrict__ outp,
                                                        int M, int N, int K) {
    __shared__ __align__(16) _Float16 sA[2][128 * 40];
    __shared__ __align__(16) _Float16 sB[2][128 * 40];

    const int m0 = blockIdx.y * 128;
    const int n0 = blockIdx.x * 128;
    const int t = threadIdx.x;
    const int wave = t >> 5, lane = t & 31;
    const int wm = (wave & 3) * 32;
    const int wn = (wave >> 2) * 64;
    const int lr = lane & 15;
    const int kb = (lane >> 4) * 8;
    const int srow = t >> 1;        // staging row (0..127)
    const int sk8  = (t & 1) * 2;   // first of two 8-half segments
    (void)srow; (void)sk8;

    const int nk = K >> 5;
    v8f acc[2][4] = {};

    auto compute = [&](int p) {
        v16h af[2], bf[4];
#pragma unroll
        for (int i = 0; i < 2; ++i)
            af[i] = load_frag16(&sA[p][(wm + i * 16 + lr) * 40 + kb]);
#pragma unroll
        for (int j = 0; j < 4; ++j)
            bf[j] = load_frag16(&sB[p][(wn + j * 16 + lr) * 40 + kb]);
#pragma unroll
        for (int i = 0; i < 2; ++i)
#pragma unroll
            for (int j = 0; j < 4; ++j)
                acc[i][j] = __builtin_amdgcn_wmma_f32_16x16x32_f16(
                    false, af[i], false, bf[j], (short)0, acc[i][j], false, false);
    };

#if USE_TDM
    const _Float16* gA = &A[(size_t)m0 * K];
    const _Float16* gB = &W[(size_t)n0 * K];
    auto stage_tdm = [&](int kc, int p) {
        tdm_load_tile(gA + (kc << 5), &sA[p][0], K);
        tdm_load_tile(gB + (kc << 5), &sB[p][0], K);
    };
    if (wave == 0) { stage_tdm(0, 0); tdm_wait0(); }
    __syncthreads();
    for (int kc = 0; kc < nk; ++kc) {
        const int p = kc & 1;
        if (wave == 0 && kc + 1 < nk) stage_tdm(kc + 1, p ^ 1);
        compute(p);
        if (wave == 0) tdm_wait0();
        __syncthreads();
    }
#elif USE_ASYNC_LDS
    auto stage = [&](int kc, int p) {
        const int k0 = kc << 5;
#pragma unroll
        for (int ss = 0; ss < 2; ++ss) {
            async_copy_b128(&A[(size_t)(m0 + srow) * K + k0 + (sk8 + ss) * 8],
                            &sA[p][srow * 40 + (sk8 + ss) * 8]);
            async_copy_b128(&W[(size_t)(n0 + srow) * K + k0 + (sk8 + ss) * 8],
                            &sB[p][srow * 40 + (sk8 + ss) * 8]);
        }
    };
    stage(0, 0);
    wait_async0();
    __syncthreads();
    for (int kc = 0; kc < nk; ++kc) {
        const int p = kc & 1;
        if (kc + 1 < nk) stage(kc + 1, p ^ 1);
        compute(p);
        wait_async0();
        __syncthreads();
    }
#else
    f32x4 ra[2], rb[2];
    auto fetch = [&](int kc) {
        const int k0 = kc << 5;
#pragma unroll
        for (int ss = 0; ss < 2; ++ss) {
            ra[ss] = *(const f32x4*)&A[(size_t)(m0 + srow) * K + k0 + (sk8 + ss) * 8];
            rb[ss] = *(const f32x4*)&W[(size_t)(n0 + srow) * K + k0 + (sk8 + ss) * 8];
        }
    };
    auto commit = [&](int p) {
#pragma unroll
        for (int ss = 0; ss < 2; ++ss) {
            *(f32x4*)&sA[p][srow * 40 + (sk8 + ss) * 8] = ra[ss];
            *(f32x4*)&sB[p][srow * 40 + (sk8 + ss) * 8] = rb[ss];
        }
    };
    fetch(0); commit(0);
    __syncthreads();
    for (int kc = 0; kc < nk; ++kc) {
        const int p = kc & 1;
        if (kc + 1 < nk) fetch(kc + 1);
        compute(p);
        if (kc + 1 < nk) commit(p ^ 1);
        __syncthreads();
    }
#endif

    // epilogue
#pragma unroll
    for (int j = 0; j < 4; ++j) {
        const int n = n0 + wn + j * 16 + lr;
        const float bv = bias[n];
#pragma unroll
        for (int i = 0; i < 2; ++i) {
            const int mbase = m0 + wm + i * 16 + (lane >> 4) * 8;
#pragma unroll
            for (int r = 0; r < 8; ++r) {
                float v = acc[i][j][r] + bv;
                if (RELU) v = fmaxf(v, 0.0f);
                const size_t idx = (size_t)(mbase + r) * N + n;
                if (RESID) {
                    ((float*)outp)[idx] = v + resid[idx];
                } else {
                    ((_Float16*)outp)[idx] = (_Float16)v;
                }
            }
        }
    }
}

// ---------------------------------------------------------------------------
// Flash attention on fused qkv [B*S, 3072] (cols: q|k|v, each h*64+d).
// Grid: (S/64, B*NHEAD). Block: 128 thr = 4 waves; wave handles 16 q rows.
// Key/value chunks of 32 double-buffered in LDS (V stored transposed);
// online softmax with wave32 16-lane butterfly reductions.
// ---------------------------------------------------------------------------
__global__ __launch_bounds__(128) void flash_attn_kernel(const _Float16* __restrict__ qkv,
                                                         _Float16* __restrict__ o) {
    __shared__ __align__(16) _Float16 sK[2][32 * 72];    // [key][d] pad->72
    __shared__ __align__(16) _Float16 sV[2][64 * 40];    // [d][key] pad->40
    __shared__ __align__(16) _Float16 sP[4][16 * 40];    // per-wave P tile

    const int bh = blockIdx.y;
    const int b = bh >> 4, h = bh & 15;
    const size_t rbase = (size_t)b * SEQ;
    const int wave = threadIdx.x >> 5, lane = threadIdx.x & 31;
    const int lr = lane & 15;
    const int kb = (lane >> 4) * 8;
    const int qbase = blockIdx.x * 64 + wave * 16;
    const float SCALE = 0.125f;  // 1/sqrt(64)
    const int colq = h * DK, colk = EMBED + h * DK, colv = 2 * EMBED + h * DK;

    v16h qf[2];
#pragma unroll
    for (int c = 0; c < 2; ++c)
        qf[c] = load_frag16(&qkv[(rbase + qbase + lr) * QKV3 + colq + c * 32 + kb]);

    v8f Oacc[4] = {};
    float mrow[8], lrow[8];
#pragma unroll
    for (int r = 0; r < 8; ++r) { mrow[r] = -1e30f; lrow[r] = 0.f; }

    f32x4 vreg[2];
    auto fetchV = [&](int kt) {
#pragma unroll
        for (int ss = 0; ss < 2; ++ss) {
            int s = threadIdx.x + ss * 128, row = s >> 3, sg = s & 7;
            vreg[ss] = *(const f32x4*)&qkv[(rbase + kt + row) * QKV3 + colv + sg * 8];
        }
    };
    auto commitV = [&](int p) {
#pragma unroll
        for (int ss = 0; ss < 2; ++ss) {
            int s = threadIdx.x + ss * 128, row = s >> 3, sg = s & 7;
            const _Float16* tp = (const _Float16*)&vreg[ss];
#pragma unroll
            for (int e = 0; e < 8; ++e) sV[p][(sg * 8 + e) * 40 + row] = tp[e];
        }
    };

    auto compute = [&](int p) {
        v8f s0 = {}, s1 = {};
#pragma unroll
        for (int c = 0; c < 2; ++c) {
            v16h kf0 = load_frag16(&sK[p][(0 + lr) * 72 + c * 32 + kb]);
            v16h kf1 = load_frag16(&sK[p][(16 + lr) * 72 + c * 32 + kb]);
            s0 = __builtin_amdgcn_wmma_f32_16x16x32_f16(false, qf[c], false, kf0,
                                                        (short)0, s0, false, false);
            s1 = __builtin_amdgcn_wmma_f32_16x16x32_f16(false, qf[c], false, kf1,
                                                        (short)0, s1, false, false);
        }
        _Float16* pw = &sP[wave][0];
#pragma unroll
        for (int r = 0; r < 8; ++r) {
            float a0 = s0[r] * SCALE;
            float a1 = s1[r] * SCALE;
            float mx = fmaxf(a0, a1);
#pragma unroll
            for (int off = 1; off < 16; off <<= 1) mx = fmaxf(mx, __shfl_xor(mx, off, 32));
            float nm = fmaxf(mrow[r], mx);
            float alpha = __expf(mrow[r] - nm);
            float p0 = __expf(a0 - nm);
            float p1 = __expf(a1 - nm);
            float rs = p0 + p1;
#pragma unroll
            for (int off = 1; off < 16; off <<= 1) rs += __shfl_xor(rs, off, 32);
            lrow[r] = lrow[r] * alpha + rs;
            mrow[r] = nm;
#pragma unroll
            for (int j = 0; j < 4; ++j) Oacc[j][r] *= alpha;
            const int prow = r + (lane >> 4) * 8;
            pw[prow * 40 + lr]      = (_Float16)p0;
            pw[prow * 40 + 16 + lr] = (_Float16)p1;
        }
        v16h pf = load_frag16(&pw[lr * 40 + kb]);     // same-wave DS: in order
#pragma unroll
        for (int j = 0; j < 4; ++j) {
            v16h vf = load_frag16(&sV[p][(j * 16 + lr) * 40 + kb]);
            Oacc[j] = __builtin_amdgcn_wmma_f32_16x16x32_f16(false, pf, false, vf,
                                                             (short)0, Oacc[j], false, false);
        }
    };

    const int nkt = SEQ / 32;
#if USE_ASYNC_LDS
    auto stageK = [&](int kt, int p) {
#pragma unroll
        for (int ss = 0; ss < 2; ++ss) {
            int s = threadIdx.x + ss * 128, row = s >> 3, sg = s & 7;
            async_copy_b128(&qkv[(rbase + kt + row) * QKV3 + colk + sg * 8],
                            &sK[p][row * 72 + sg * 8]);
        }
    };
    stageK(0, 0); fetchV(0); commitV(0);
    wait_async0();
    __syncthreads();
    for (int i = 0; i < nkt; ++i) {
        const int p = i & 1;
        if (i + 1 < nkt) { stageK((i + 1) * 32, p ^ 1); fetchV((i + 1) * 32); }
        compute(p);
        if (i + 1 < nkt) commitV(p ^ 1);
        wait_async0();
        __syncthreads();
    }
#else
    f32x4 kreg[2];
    auto fetchK = [&](int kt) {
#pragma unroll
        for (int ss = 0; ss < 2; ++ss) {
            int s = threadIdx.x + ss * 128, row = s >> 3, sg = s & 7;
            kreg[ss] = *(const f32x4*)&qkv[(rbase + kt + row) * QKV3 + colk + sg * 8];
        }
    };
    auto commitK = [&](int p) {
#pragma unroll
        for (int ss = 0; ss < 2; ++ss) {
            int s = threadIdx.x + ss * 128, row = s >> 3, sg = s & 7;
            *(f32x4*)&sK[p][row * 72 + sg * 8] = kreg[ss];
        }
    };
    fetchK(0); fetchV(0); commitK(0); commitV(0);
    __syncthreads();
    for (int i = 0; i < nkt; ++i) {
        const int p = i & 1;
        if (i + 1 < nkt) { fetchK((i + 1) * 32); fetchV((i + 1) * 32); }
        compute(p);
        if (i + 1 < nkt) { commitK(p ^ 1); commitV(p ^ 1); }
        __syncthreads();
    }
#endif

    // epilogue: normalize by row sum, write f16 at stride EMBED
#pragma unroll
    for (int j = 0; j < 4; ++j) {
#pragma unroll
        for (int r = 0; r < 8; ++r) {
            float val = Oacc[j][r] / lrow[r];
            o[(rbase + qbase + r + (lane >> 4) * 8) * EMBED + h * DK + j * 16 + lr] =
                (_Float16)val;
        }
    }
}

// ---------------------------------------------------------------------------
// Host-side orchestration
// ---------------------------------------------------------------------------
extern "C" void kernel_launch(void* const* d_in, const int* in_sizes, int n_in,
                              void* d_out, int out_size, void* d_ws, size_t ws_size,
                              hipStream_t stream) {
    (void)in_sizes; (void)n_in; (void)out_size; (void)ws_size;
    const float* x   = (const float*)d_in[0];
    const float* Wq  = (const float*)d_in[1];  const float* bq  = (const float*)d_in[2];
    const float* Wk  = (const float*)d_in[3];  const float* bk  = (const float*)d_in[4];
    const float* Wv  = (const float*)d_in[5];  const float* bv  = (const float*)d_in[6];
    const float* Wo  = (const float*)d_in[7];  const float* bo  = (const float*)d_in[8];
    const float* W1  = (const float*)d_in[9];  const float* b1  = (const float*)d_in[10];
    const float* W2  = (const float*)d_in[11]; const float* b2  = (const float*)d_in[12];
    const float* g1  = (const float*)d_in[13]; const float* be1 = (const float*)d_in[14];
    const float* g2  = (const float*)d_in[15]; const float* be2 = (const float*)d_in[16];
    float* out = (float*)d_out;

    // workspace carve-up (256B aligned)
    char* w = (char*)d_ws;
    auto carve = [&](size_t bytes) { char* p = w; w += (bytes + 255) & ~(size_t)255; return p; };
    _Float16* hF    = (_Float16*)carve((size_t)MTOT * EMBED * 2);
    _Float16* qkvF  = (_Float16*)carve((size_t)MTOT * QKV3 * 2);
    _Float16* aF    = (_Float16*)carve((size_t)MTOT * EMBED * 2);
    _Float16* h2F   = (_Float16*)carve((size_t)MTOT * EMBED * 2);
    _Float16* ffnF  = (_Float16*)carve((size_t)MTOT * FFN * 2);
    float*    x1    = (float*)   carve((size_t)MTOT * EMBED * 4);
    _Float16* WqkvF = (_Float16*)carve((size_t)QKV3 * EMBED * 2);
    _Float16* WoF   = (_Float16*)carve((size_t)EMBED * EMBED * 2);
    _Float16* W1F   = (_Float16*)carve((size_t)FFN * EMBED * 2);
    _Float16* W2F   = (_Float16*)carve((size_t)EMBED * FFN * 2);
    float*    bqkv  = (float*)   carve((size_t)QKV3 * 4);

    // 1) weight conversion fp32 -> fp16 (QKV fused into one [3072,1024] buffer)
    const int nEE = EMBED * EMBED, nFE = FFN * EMBED;
    f32_to_f16_kernel<<<nEE / 256, 256, 0, stream>>>(Wq, WqkvF,           nEE);
    f32_to_f16_kernel<<<nEE / 256, 256, 0, stream>>>(Wk, WqkvF + nEE,     nEE);
    f32_to_f16_kernel<<<nEE / 256, 256, 0, stream>>>(Wv, WqkvF + 2 * nEE, nEE);
    f32_to_f16_kernel<<<nEE / 256, 256, 0, stream>>>(Wo, WoF, nEE);
    f32_to_f16_kernel<<<nFE / 256, 256, 0, stream>>>(W1, W1F, nFE);
    f32_to_f16_kernel<<<nFE / 256, 256, 0, stream>>>(W2, W2F, nFE);
    concat_bias_kernel<<<QKV3 / 256, 256, 0, stream>>>(bq, bk, bv, bqkv);

    // 2) LayerNorm 1
    layernorm_f16_kernel<<<MTOT, 256, 0, stream>>>(x, g1, be1, hF);

    // 3) fused QKV projection: [4096,1024] x [3072,1024]^T -> [4096,3072]
    dim3 gQKV(QKV3 / 128, MTOT / 128);   // (24, 32)
    wmma_gemm_kernel<false, false><<<gQKV, 256, 0, stream>>>(hF, WqkvF, bqkv, nullptr, qkvF,
                                                             MTOT, QKV3, EMBED);

    // 4) attention
    dim3 gAttn(SEQ / 64, 2 * NHEAD);     // (32, 32)
    flash_attn_kernel<<<gAttn, 128, 0, stream>>>(qkvF, aF);

    // 5) output projection + residual (fp32)
    dim3 gEE(EMBED / 128, MTOT / 128);   // (8, 32)
    wmma_gemm_kernel<false, true><<<gEE, 256, 0, stream>>>(aF, WoF, bo, x, x1,
                                                           MTOT, EMBED, EMBED);

    // 6) LayerNorm 2
    layernorm_f16_kernel<<<MTOT, 256, 0, stream>>>(x1, g2, be2, h2F);

    // 7) FFN up + ReLU
    dim3 gFE(FFN / 128, MTOT / 128);     // (32, 32)
    wmma_gemm_kernel<true, false><<<gFE, 256, 0, stream>>>(h2F, W1F, b1, nullptr, ffnF,
                                                           MTOT, FFN, EMBED);

    // 8) FFN down + residual -> output (fp32)
    wmma_gemm_kernel<false, true><<<gEE, 256, 0, stream>>>(ffnF, W2F, b2, x1, out,
                                                           MTOT, EMBED, FFN);
}
